// NERModel_5918464934521
// MI455X (gfx1250) — compile-verified, matrix operand order checked by reading, not run on hardware
//
#include <hip/hip_runtime.h>
#include <stdint.h>

#define SEQ   4096
#define EMB   300
#define HID   500
#define NCLS  20
#define H4    2000          // 4*HID
#define KPAD  608           // 2*EMB padded to multiple of 32 (19 k-tiles)
#define HPAD  512           // HID padded to multiple of 32
#define NWG_SCAN 20
#define UPW   25            // hidden units per scan workgroup (20*25 = 500)
#define GROWS 100           // 4*UPW gate rows per scan workgroup

typedef __attribute__((ext_vector_type(16))) __bf16 v16bf;
typedef __attribute__((ext_vector_type(8)))  float  v8f;

__device__ __forceinline__ unsigned short f2bf(float f) {
  unsigned u = __builtin_bit_cast(unsigned, f);
  unsigned r = 0x7FFFu + ((u >> 16) & 1u);   // round-to-nearest-even
  return (unsigned short)((u + r) >> 16);
}
__device__ __forceinline__ float bf2f(unsigned short b) {
  return __builtin_bit_cast(float, ((unsigned)b) << 16);
}
__device__ __forceinline__ float sigmoidf(float x) {
  return 1.0f / (1.0f + __expf(-x));
}

// ---------------- prep kernels: fp32 -> bf16 packing (K padded) ----------------
__global__ void k_pack_wih(const float* __restrict__ w, unsigned short* __restrict__ wb) {
  int idx = blockIdx.x * blockDim.x + threadIdx.x;
  if (idx >= H4 * KPAD) return;
  int n = idx / KPAD, k = idx % KPAD;
  wb[idx] = (k < 2 * EMB) ? f2bf(w[(size_t)n * (2 * EMB) + k]) : (unsigned short)0;
}

__global__ void k_pack_whh(const float* __restrict__ w, unsigned short* __restrict__ wb) {
  int idx = blockIdx.x * blockDim.x + threadIdx.x;
  if (idx >= H4 * HPAD) return;
  int n = idx / HPAD, k = idx % HPAD;
  wb[idx] = (k < HID) ? f2bf(w[(size_t)n * HID + k]) : (unsigned short)0;
}

// embedding gather + concat -> bf16 x [SEQ][KPAD]
__global__ void k_gather(const int* __restrict__ words, const int* __restrict__ labels,
                         const float* __restrict__ ew, const float* __restrict__ ee,
                         unsigned short* __restrict__ xb) {
  int idx = blockIdx.x * blockDim.x + threadIdx.x;
  if (idx >= SEQ * KPAD) return;
  int t = idx / KPAD, k = idx % KPAD;
  float v = 0.0f;
  if (k < EMB)            v = ew[(size_t)words[t]  * EMB + k];
  else if (k < 2 * EMB)   v = ee[(size_t)labels[t] * EMB + (k - EMB)];
  xb[idx] = f2bf(v);
}

__global__ void k_bias(const float* __restrict__ bih, const float* __restrict__ bhh,
                       float* __restrict__ bias, unsigned* __restrict__ cnt) {
  int i = blockIdx.x * blockDim.x + threadIdx.x;
  if (i < H4) bias[i] = bih[i] + bhh[i];
  if (i == 0) *cnt = 0u;   // reset barrier counter every call (deterministic)
}

// ---------------- phase 1: pre = x @ W_ih^T + b via WMMA bf16 ----------------
// One wave per 16x16 output tile. A = xb[SEQ][KPAD], B = W_ihb[H4][KPAD]
// (both row-major with K contiguous -> identical fragment load pattern).
// 16-bit fragment layout (ISA 7.12.2): lane L holds M(or N)=L%16;
// pair p (VGPR p) holds K = (L>=16?8:0) + (p>=4?16:0) + 2*(p&3) .. +1
__global__ void __launch_bounds__(256)
k_gemm(const unsigned short* __restrict__ xb, const unsigned short* __restrict__ wb,
       const float* __restrict__ bias, float* __restrict__ pre) {
  const int NT = H4 / 16;                                    // 125 n-tiles
  const int wave = blockIdx.x * (blockDim.x >> 5) + (threadIdx.x >> 5);
  if (wave >= (SEQ / 16) * NT) return;                       // wave-uniform: EXEC stays full
  const int mt = wave / NT, nt = wave % NT;
  const int lane = threadIdx.x & 31;
  const int half = lane >> 4, l16 = lane & 15;
  const unsigned short* arow = xb + (size_t)(mt * 16 + l16) * KPAD;
  const unsigned short* brow = wb + (size_t)(nt * 16 + l16) * KPAD;

  v8f acc = {};
#pragma unroll 4
  for (int kk = 0; kk < KPAD; kk += 32) {
    __builtin_prefetch(arow + kk + 128, 0, 1);   // global_prefetch_b8
    __builtin_prefetch(brow + kk + 128, 0, 1);
    union { unsigned u[8]; v16bf v; } A, B;
#pragma unroll
    for (int p = 0; p < 8; ++p) {
      const int K = kk + (half << 3) + ((p >= 4) ? 16 : 0) + ((p & 3) << 1);
      A.u[p] = *(const unsigned*)(arow + K);     // two packed bf16, dword-aligned
      B.u[p] = *(const unsigned*)(brow + K);
    }
    acc = __builtin_amdgcn_wmma_f32_16x16x32_bf16(
        /*neg_a=*/false, A.v, /*neg_b=*/false, B.v,
        /*c_mod=*/(short)0, acc, /*reuse_a=*/false, /*reuse_b=*/false);
  }
  // C/D layout: VGPR r -> M = r + (lane<16 ? 0 : 8), N = lane%16
  const int n  = nt * 16 + l16;
  const float bv = bias[n];
#pragma unroll
  for (int r = 0; r < 8; ++r) {
    const int m = mt * 16 + r + (half << 3);
    pre[(size_t)m * H4 + n] = acc[r] + bv;
  }
}

// ---------------- phase 2: persistent LSTM scan, 20 cooperating WGs ----------------
// WG w owns hidden units [w*25, w*25+25); its 100 gate rows of W_hh live in LDS
// (100*512*2B = 100 KB of the WGP's 320 KB). Per step: LDS-resident matvec,
// local c/h update, h-slice exchange through L2 with a monotonic atomic barrier.
__global__ void __launch_bounds__(256)
k_scan(const float* __restrict__ pre, const unsigned short* __restrict__ whh,
       float* __restrict__ hbuf, unsigned* __restrict__ cnt) {
  extern __shared__ char smem[];
  unsigned short* wlds = (unsigned short*)smem;              // [GROWS][HPAD] bf16
  float* h_s  = (float*)(smem + (size_t)GROWS * HPAD * 2);   // [HPAD]
  float* gbuf = h_s + HPAD;                                  // [GROWS]
  float* cbuf = gbuf + GROWS;                                // [UPW]

  const int tid = threadIdx.x, wg = blockIdx.x;
  const int lane = tid & 31, wv = tid >> 5;                  // 8 waves of 32

  // stage this WG's W_hh rows into LDS (dword copies)
  for (int idx = tid; idx < GROWS * (HPAD / 2); idx += blockDim.x) {
    int r  = idx >> 8;            // / (HPAD/2 == 256)
    int kp = idx & 255;
    int gr = (r / UPW) * HID + wg * UPW + (r % UPW);         // i,f,g,o row in W_hh
    ((unsigned*)wlds)[idx] = ((const unsigned*)whh)[(size_t)gr * (HPAD / 2) + kp];
  }
  for (int k = tid; k < HPAD; k += blockDim.x) h_s[k] = 0.0f;
  if (tid < UPW) cbuf[tid] = 0.0f;
  __syncthreads();

  for (int t = 0; t < SEQ; ++t) {
    // gates[r] = pre[t][gr] + dot(W_hh[gr], h) : one wave per row, 8 rows at a time
    for (int r = wv; r < GROWS; r += 8) {
      const unsigned short* wr = wlds + (size_t)r * HPAD;
      float acc = 0.0f;
#pragma unroll
      for (int k = lane; k < HPAD; k += 32)
        acc += bf2f(wr[k]) * h_s[k];
      for (int off = 16; off > 0; off >>= 1)
        acc += __shfl_down(acc, off, 32);
      if (lane == 0) {
        int gr = (r / UPW) * HID + wg * UPW + (r % UPW);
        gbuf[r] = acc + pre[(size_t)t * H4 + gr];
      }
    }
    __syncthreads();
    if (tid < UPW) {
      float gi = sigmoidf(gbuf[tid]);
      float gf = sigmoidf(gbuf[UPW + tid]);
      float gg = tanhf   (gbuf[2 * UPW + tid]);
      float go = sigmoidf(gbuf[3 * UPW + tid]);
      float c  = gf * cbuf[tid] + gi * gg;
      cbuf[tid] = c;
      hbuf[wg * UPW + tid] = go * tanhf(c);   // publish h slice
      __threadfence();
    }
    __syncthreads();
    if (tid == 0) {   // monotonic device-scope arrive + spin
      __hip_atomic_fetch_add(cnt, 1u, __ATOMIC_ACQ_REL, __HIP_MEMORY_SCOPE_AGENT);
      const unsigned tgt = (unsigned)(t + 1) * NWG_SCAN;
      while (__hip_atomic_load(cnt, __ATOMIC_ACQUIRE, __HIP_MEMORY_SCOPE_AGENT) < tgt)
        __builtin_amdgcn_s_sleep(1);
      __threadfence();
    }
    __syncthreads();
    for (int k = tid; k < HID; k += blockDim.x) h_s[k] = hbuf[k];   // h for next step
    __syncthreads();
  }
}

// ---------------- phase 3: out = h_last @ fc_w^T + fc_b ----------------
__global__ void k_fc(const float* __restrict__ hbuf, const float* __restrict__ fcw,
                     const float* __restrict__ fcb, float* __restrict__ out) {
  const int lane = threadIdx.x & 31, wv = threadIdx.x >> 5;
  for (int c = wv; c < NCLS; c += (blockDim.x >> 5)) {
    float acc = 0.0f;
    for (int k = lane; k < HID; k += 32)
      acc += fcw[(size_t)c * HID + k] * hbuf[k];
    for (int off = 16; off > 0; off >>= 1)
      acc += __shfl_down(acc, off, 32);
    if (lane == 0) out[c] = acc + fcb[c];
  }
}

extern "C" void kernel_launch(void* const* d_in, const int* in_sizes, int n_in,
                              void* d_out, int out_size, void* d_ws, size_t ws_size,
                              hipStream_t stream) {
  (void)in_sizes; (void)n_in; (void)out_size; (void)ws_size;
  const int*   words  = (const int*)  d_in[0];
  const int*   labels = (const int*)  d_in[1];
  const float* ew     = (const float*)d_in[2];
  const float* ee     = (const float*)d_in[3];
  const float* wih    = (const float*)d_in[4];
  const float* whh    = (const float*)d_in[5];
  const float* bih    = (const float*)d_in[6];
  const float* bhh    = (const float*)d_in[7];
  const float* fcw    = (const float*)d_in[8];
  const float* fcb    = (const float*)d_in[9];
  float* out = (float*)d_out;

  char* ws = (char*)d_ws;
  size_t off = 0;
  auto alloc = [&](size_t bytes) {
    char* p = ws + off;
    off = (off + bytes + 255) & ~(size_t)255;
    return p;
  };
  unsigned short* xb   = (unsigned short*)alloc((size_t)SEQ * KPAD * 2);  // ~5.0 MB
  unsigned short* wihb = (unsigned short*)alloc((size_t)H4  * KPAD * 2);  // ~2.4 MB
  unsigned short* whhb = (unsigned short*)alloc((size_t)H4  * HPAD * 2);  // ~2.0 MB
  float*    pre  = (float*)alloc((size_t)SEQ * H4 * 4);                   // ~32.8 MB
  float*    bias = (float*)alloc((size_t)H4 * 4);
  float*    hbuf = (float*)alloc((size_t)HPAD * 4);
  unsigned* cnt  = (unsigned*)alloc(256);

  dim3 b(256);
  k_pack_wih<<<(H4 * KPAD + 255) / 256, b, 0, stream>>>(wih, wihb);
  k_pack_whh<<<(H4 * HPAD + 255) / 256, b, 0, stream>>>(whh, whhb);
  k_gather  <<<(SEQ * KPAD + 255) / 256, b, 0, stream>>>(words, labels, ew, ee, xb);
  k_bias    <<<(H4 + 255) / 256, b, 0, stream>>>(bih, bhh, bias, cnt);

  // 32000 tiles of 16x16, 8 waves per block -> 4000 blocks
  k_gemm<<<(SEQ / 16) * (H4 / 16) / 8, b, 0, stream>>>(xb, wihb, bias, pre);

  const size_t smem = (size_t)GROWS * HPAD * 2 + (size_t)HPAD * 4
                    + (size_t)GROWS * 4 + (size_t)UPW * 4;   // ~105 KB of 320 KB LDS
  k_scan<<<NWG_SCAN, b, smem, stream>>>(pre, whhb, hbuf, cnt);

  k_fc<<<1, b, 0, stream>>>(hbuf, fcw, fcb, out);
}
